// RNN_7842610283034
// MI455X (gfx1250) — matrix operand order repaired and checked by reading, not compiled, not run on hardware
//
#include <hip/hip_runtime.h>

// ---------------------------------------------------------------------------
// RNN (relu Elman cell) for MI455X / gfx1250.
// Per-step GEMMs via V_WMMA_F32_16X16X32_BF16 (wave32, 16x16 f32 accum).
// Weights converted once fp32 -> bf16 (16MB total, resident in 192MB L2 for
// all 257 sequential steps). LDS layouts chosen so every WMMA fragment is two
// contiguous 16-byte reads -> ds_load_b128, no scalar gathers / b16 packing.
// ---------------------------------------------------------------------------

#define BATCH   256
#define WARMUPT 128
#define HID     2048
#define FEAT    1024
#define TSTEPS  128

typedef __bf16 bf16_t;
typedef __bf16 v16bf __attribute__((ext_vector_type(16)));
typedef __bf16 v8bf  __attribute__((ext_vector_type(8)));
typedef float  v8f   __attribute__((ext_vector_type(8)));

union Frag16 {
  v16bf v;
  v8bf  h[2];
};

__device__ __forceinline__ bf16_t f2bf(float f) {
  union { float f; unsigned u; } v; v.f = f;
  unsigned r = (v.u + 0x7FFFu + ((v.u >> 16) & 1u)) >> 16;  // round-nearest-even
  union { unsigned short s; bf16_t b; } o; o.s = (unsigned short)r;
  return o.b;
}

// --------------------------- helper kernels --------------------------------

__global__ __launch_bounds__(256) void cvt_f32_bf16(const float* __restrict__ src,
                                                    bf16_t* __restrict__ dst, int n) {
  int i = blockIdx.x * blockDim.x + threadIdx.x;
  int stride = gridDim.x * blockDim.x;
  for (; i < n; i += stride) dst[i] = f2bf(src[i]);
}

__global__ __launch_bounds__(256) void zero_bf16(bf16_t* __restrict__ dst, int n) {
  int i = blockIdx.x * blockDim.x + threadIdx.x;
  int stride = gridDim.x * blockDim.x;
  union { unsigned short s; bf16_t b; } z; z.s = 0;
  for (; i < n; i += stride) dst[i] = z.b;
}

// --------------------------- tiled WMMA GEMM core --------------------------
// Accumulates C[bm:bm+128, bn:bn+64] += A[.,K] * W[N,K]^T into acc[2][2].
// Block: 256 threads = 8 waves in a 4(M) x 2(N) grid; wave owns 32x32.
// K staged through LDS in chunks of 64 (two WMMA k-slices of 32).
// lA layout: [m][k]  (row m contiguous in k)  -> A frag = 2x contiguous 16B
// lB layout: [n][k]  (same as W, pure copy)   -> B frag = 2x contiguous 16B
template <bool AF32>
__device__ __forceinline__ void gemm_tiles(
    const void* __restrict__ Aptr, long lda, int K,
    const bf16_t* __restrict__ W,        // [N, K] row-major bf16
    int bm, int bn, int wm, int wn, int lane,
    bf16_t* lA, bf16_t* lB, v8f acc[2][2]) {
  const int tid = threadIdx.x;
  const int l15 = lane & 15;
  const int lhi = lane >> 4;

  for (int k0 = 0; k0 < K; k0 += 64) {
    __syncthreads();
    // ---- stage A tile: 128 rows x 64 k
    if (AF32) {
      const float* X = (const float*)Aptr;
#pragma unroll
      for (int i = 0; i < 8; ++i) {
        int v = tid + i * 256;            // 2048 float4's
        int row = (v * 4) >> 6;
        int col = (v * 4) & 63;
        float4 f = *(const float4*)(X + (long)(bm + row) * lda + k0 + col);
        bf16_t* p = lA + row * 64 + col;
        p[0] = f2bf(f.x); p[1] = f2bf(f.y); p[2] = f2bf(f.z); p[3] = f2bf(f.w);
      }
    } else {
      const bf16_t* X = (const bf16_t*)Aptr;
#pragma unroll
      for (int i = 0; i < 4; ++i) {
        int v = tid + i * 256;            // 1024 v8bf's
        int row = v >> 3;
        int col = (v & 7) * 8;
        *(v8bf*)(lA + row * 64 + col) =
            *(const v8bf*)(X + (long)(bm + row) * lda + k0 + col);
      }
    }
    // ---- stage B tile: straight [n][k] copy of W rows (contiguous 16B both ends)
#pragma unroll
    for (int i = 0; i < 2; ++i) {
      int v = tid + i * 256;              // 512 v8bf's
      int n = v >> 3;
      int k = (v & 7) * 8;
      *(v8bf*)(lB + n * 64 + k) = *(const v8bf*)(W + (long)(bn + n) * K + k0 + k);
    }
    __syncthreads();

    // ---- two WMMA k-slices of 32
#pragma unroll
    for (int ks = 0; ks < 64; ks += 32) {
      Frag16 af[2], bf[2];
      // A frag (ISA 7.12.2, 16-bit A 16x32): lanes 0-15 K={0..7,16..23},
      // lanes 16-31 K={8..15,24..31}; both halves contiguous 8-element runs.
#pragma unroll
      for (int mt = 0; mt < 2; ++mt) {
        const bf16_t* base = lA + (wm + mt * 16 + l15) * 64 + ks + lhi * 8;
        af[mt].h[0] = *(const v8bf*)(base);
        af[mt].h[1] = *(const v8bf*)(base + 16);
      }
      // B frag (32x16): lane holds column n = lane&15, lanes 0-15 K={0..7,16..23},
      // lanes 16-31 K={8..15,24..31} (same k order as A); two contiguous 16B runs.
#pragma unroll
      for (int nt = 0; nt < 2; ++nt) {
        const bf16_t* base = lB + (wn + nt * 16 + l15) * 64 + ks + lhi * 8;
        bf[nt].h[0] = *(const v8bf*)(base);
        bf[nt].h[1] = *(const v8bf*)(base + 16);
      }
#pragma unroll
      for (int mt = 0; mt < 2; ++mt)
#pragma unroll
        for (int nt = 0; nt < 2; ++nt)
          acc[mt][nt] = __builtin_amdgcn_wmma_f32_16x16x32_bf16(
              false, af[mt].v, false, bf[nt].v, (short)0, acc[mt][nt], false, false);
    }
  }
}

// --------------------------- fused RNN cell --------------------------------
// h' = relu(X @ Wih^T + Hin @ Whh^T + bih + bhh), written as bf16.
template <bool AF32>
__global__ __launch_bounds__(256) void cell_kernel(
    const void* __restrict__ X, long ldx,
    const bf16_t* __restrict__ Wih,
    const bf16_t* __restrict__ Hin,
    const bf16_t* __restrict__ Whh,
    const float* __restrict__ bih,
    const float* __restrict__ bhh,
    bf16_t* __restrict__ Hout) {
  __shared__ bf16_t lA[128 * 64];
  __shared__ bf16_t lB[64 * 64];
  const int bm = blockIdx.x * 128;
  const int bn = blockIdx.y * 64;
  const int w = threadIdx.x >> 5;
  const int lane = threadIdx.x & 31;
  const int wm = (w & 3) * 32;
  const int wn = (w >> 2) * 32;

  v8f acc[2][2] = {};
  gemm_tiles<AF32>(X, ldx, FEAT, Wih, bm, bn, wm, wn, lane, lA, lB, acc);
  gemm_tiles<false>(Hin, (long)HID, HID, Whh, bm, bn, wm, wn, lane, lA, lB, acc);

  const int l15 = lane & 15, lhi = lane >> 4;
#pragma unroll
  for (int mt = 0; mt < 2; ++mt)
#pragma unroll
    for (int nt = 0; nt < 2; ++nt) {
      int gm = bm + wm + mt * 16 + lhi * 8;   // C layout: VGPR r -> M = r + 8*(lane>=16)
      int gn = bn + wn + nt * 16 + l15;       // N = lane & 15
      float bias = bih[gn] + bhh[gn];
#pragma unroll
      for (int r = 0; r < 8; ++r) {
        float vv = acc[mt][nt][r] + bias;
        vv = vv > 0.f ? vv : 0.f;
        Hout[(long)(gm + r) * HID + gn] = f2bf(vv);
      }
    }
}

// --------------------------- fc / prediction -------------------------------
// pred = H @ Wfc^T + bfc ; fp32 to d_out[:, tstep, :], bf16 to feedback buffer.
__global__ __launch_bounds__(256) void fc_kernel(
    const bf16_t* __restrict__ H,
    const bf16_t* __restrict__ Wfc,
    const float* __restrict__ bfc,
    float* __restrict__ out, int tstep,
    bf16_t* __restrict__ pred) {
  __shared__ bf16_t lA[128 * 64];
  __shared__ bf16_t lB[64 * 64];
  const int bm = blockIdx.x * 128;
  const int bn = blockIdx.y * 64;
  const int w = threadIdx.x >> 5;
  const int lane = threadIdx.x & 31;
  const int wm = (w & 3) * 32;
  const int wn = (w >> 2) * 32;

  v8f acc[2][2] = {};
  gemm_tiles<false>(H, (long)HID, HID, Wfc, bm, bn, wm, wn, lane, lA, lB, acc);

  const int l15 = lane & 15, lhi = lane >> 4;
#pragma unroll
  for (int mt = 0; mt < 2; ++mt)
#pragma unroll
    for (int nt = 0; nt < 2; ++nt) {
      int gm = bm + wm + mt * 16 + lhi * 8;
      int gn = bn + wn + nt * 16 + l15;
      float bias = bfc[gn];
#pragma unroll
      for (int r = 0; r < 8; ++r) {
        float vv = acc[mt][nt][r] + bias;
        out[((long)(gm + r) * (TSTEPS + 1) + tstep) * FEAT + gn] = vv;
        pred[(long)(gm + r) * FEAT + gn] = f2bf(vv);
      }
    }
}

// --------------------------- host driver -----------------------------------

extern "C" void kernel_launch(void* const* d_in, const int* in_sizes, int n_in,
                              void* d_out, int out_size, void* d_ws, size_t ws_size,
                              hipStream_t stream) {
  const float* inputs = (const float*)d_in[0];
  const float* W_ih = (const float*)d_in[1];
  const float* b_ih = (const float*)d_in[2];
  const float* W_hh = (const float*)d_in[3];
  const float* b_hh = (const float*)d_in[4];
  const float* W_fc = (const float*)d_in[5];
  const float* b_fc = (const float*)d_in[6];
  float* out = (float*)d_out;

  // workspace carve (~18.5 MB)
  char* ws = (char*)d_ws;
  bf16_t* wih = (bf16_t*)ws;  ws += (size_t)HID * FEAT * sizeof(bf16_t);
  bf16_t* whh = (bf16_t*)ws;  ws += (size_t)HID * HID * sizeof(bf16_t);
  bf16_t* wfc = (bf16_t*)ws;  ws += (size_t)FEAT * HID * sizeof(bf16_t);
  bf16_t* hA  = (bf16_t*)ws;  ws += (size_t)BATCH * HID * sizeof(bf16_t);
  bf16_t* hB  = (bf16_t*)ws;  ws += (size_t)BATCH * HID * sizeof(bf16_t);
  bf16_t* pred = (bf16_t*)ws;

  cvt_f32_bf16<<<512, 256, 0, stream>>>(W_ih, wih, HID * FEAT);
  cvt_f32_bf16<<<1024, 256, 0, stream>>>(W_hh, whh, HID * HID);
  cvt_f32_bf16<<<512, 256, 0, stream>>>(W_fc, wfc, FEAT * HID);
  zero_bf16<<<256, 256, 0, stream>>>(hA, BATCH * HID);

  dim3 gridCell(BATCH / 128, HID / 64);   // 2 x 32 = 64 blocks
  dim3 gridFc(BATCH / 128, FEAT / 64);    // 2 x 16 = 32 blocks

  bf16_t* hin = hA;
  bf16_t* hout = hB;

  // warmup: x_t is a strided fp32 slice of inputs [B, WARMUP, FEAT]
  for (int t = 0; t < WARMUPT; ++t) {
    cell_kernel<true><<<gridCell, 256, 0, stream>>>(
        (const void*)(inputs + (long)t * FEAT), (long)WARMUPT * FEAT,
        wih, hin, whh, b_ih, b_hh, hout);
    bf16_t* tmp = hin; hin = hout; hout = tmp;
  }

  fc_kernel<<<gridFc, 256, 0, stream>>>(hin, wfc, b_fc, out, 0, pred);

  // autoregressive rollout: feed bf16 prediction back as input
  for (int t = 0; t < TSTEPS; ++t) {
    cell_kernel<false><<<gridCell, 256, 0, stream>>>(
        (const void*)pred, (long)FEAT, wih, hin, whh, b_ih, b_hh, hout);
    bf16_t* tmp = hin; hin = hout; hout = tmp;
    fc_kernel<<<gridFc, 256, 0, stream>>>(hin, wfc, b_fc, out, t + 1, pred);
  }
}
